// AttentionHead_5961414607077
// MI455X (gfx1250) — compile-verified, matrix operand order checked by reading, not run on hardware
//
#include <hip/hip_runtime.h>
#include <math.h>

// ---------------------------------------------------------------------------
// Causal single-head attention for MI455X (gfx1250, wave32, WMMA, TDM).
//   N_CTX=8192, D_MODEL=1024, D_HEAD=128, fp32 I/O, bf16 WMMA internally.
// Workspace layout (bf16): Qb | Kb | Vb | Ob, each 8192*128 elements (8 MB).
// ---------------------------------------------------------------------------

#define N_CTX   8192
#define D_MODEL 1024
#define D_HEAD  128

typedef __attribute__((ext_vector_type(4)))  float  v4f;
typedef __attribute__((ext_vector_type(8)))  float  v8f;
typedef __attribute__((ext_vector_type(4)))  __bf16 v4bf;
typedef __attribute__((ext_vector_type(8)))  __bf16 v8bf;
typedef __attribute__((ext_vector_type(16))) __bf16 v16bf;
typedef unsigned int u32x4 __attribute__((ext_vector_type(4)));
typedef int          i32x4 __attribute__((ext_vector_type(4)));
typedef int          i32x8 __attribute__((ext_vector_type(8)));

#if defined(__has_builtin)
#if __has_builtin(__builtin_amdgcn_tensor_load_to_lds) && \
    __has_builtin(__builtin_amdgcn_s_wait_tensorcnt)
#define USE_TDM 1
#endif
#if __has_builtin(__builtin_amdgcn_sched_group_barrier)
#define SCHED_DS_WMMA(nds, nmma)                                     \
  do {                                                               \
    __builtin_amdgcn_sched_group_barrier(0x100, (nds), 0);  /* DS read */ \
    __builtin_amdgcn_sched_group_barrier(0x008, (nmma), 0); /* MFMA/WMMA */ \
  } while (0)
#endif
#endif
#ifndef USE_TDM
#define USE_TDM 0
#endif
#ifndef SCHED_DS_WMMA
#define SCHED_DS_WMMA(nds, nmma) do {} while (0)
#endif

// 16x16x32 bf16 WMMA, fp32 accumulate.
static __device__ inline v8f wmma_bf16(v16bf a, v16bf b, v8f c) {
  return __builtin_amdgcn_wmma_f32_16x16x32_bf16(
      /*neg_a=*/false, a, /*neg_b=*/false, b,
      /*c_mod=*/(short)0, c, /*reuse_a=*/false, /*reuse_b=*/false);
}

// Load a 16x32 bf16 fragment (A-operand layout, ISA 7.12.2) from a row-major
// bf16 array: two contiguous 16-byte chunks per lane (K=[8h,8h+8) and
// [16+8h,16+8h+8), h = lane>>4) -> ds_load_b128 / global_load_b128.
// B operands use the same per-lane pattern indexed by (N,K), so staging the
// B matrix as [n][k] lets this loader serve both operands.
static __device__ inline v16bf load_frag16(const __bf16* __restrict__ base,
                                           int row0, int k0, int stride) {
  const int lane = threadIdx.x & 31;
  const int hlf  = lane >> 4;
  const int rr   = lane & 15;
  const __bf16* p = base + (row0 + rr) * stride + k0 + 8 * hlf;
  v8bf c0 = *(const v8bf*)(p);
  v8bf c1 = *(const v8bf*)(p + 16);
  v16bf out;
#pragma unroll
  for (int i = 0; i < 8; ++i) { out[i] = c0[i]; out[8 + i] = c1[i]; }
  return out;
}

static __device__ inline v8f v8f_zero() {
  v8f z;
#pragma unroll
  for (int i = 0; i < 8; ++i) z[i] = 0.0f;
  return z;
}

// ---------------------------------------------------------------------------
// Kernel 1: Q/K/V projection.  x[8192,1024] @ W[1024,128] -> bf16 [8192,128].
// grid = (64, 3); block = 256 (8 waves). Double-buffered 64-wide K slabs.
// launch_bounds(.,1): allow a fat register budget so the 8 B fragments stay
// resident and WMMAs issue as a burst behind one dscnt wait.
// ---------------------------------------------------------------------------
__global__ __launch_bounds__(256, 1) void qkv_proj_kernel(
    const float* __restrict__ x,  const float* __restrict__ Wq,
    const float* __restrict__ Wk, const float* __restrict__ Wv,
    __bf16* __restrict__ Qb, __bf16* __restrict__ Kb, __bf16* __restrict__ Vb)
{
  __shared__ alignas(16) __bf16 xs[2][128 * 64];   // [m][k]   2 x 16 KB
  __shared__ alignas(16) __bf16 wt[2][128 * 64];   // [n][k]   2 x 16 KB

  const int which = blockIdx.y;
  const float* __restrict__ W = (which == 0) ? Wq : (which == 1) ? Wk : Wv;
  __bf16* __restrict__ dst    = (which == 0) ? Qb : (which == 1) ? Kb : Vb;
  const float scale = (which == 0) ? 0.08838834764831845f : 1.0f; // 1/sqrt(128)

  const int mBase = blockIdx.x * 128;
  const int tid  = threadIdx.x;
  const int wave = tid >> 5;
  const int lane = tid & 31;
  const int hlf  = lane >> 4;
  const int rr   = lane & 15;

  auto stage = [&](int kBase, int buf) {
    // x slab: float4 reads, v4bf LDS writes (coalesced).
    for (int i = tid; i < 128 * 16; i += 256) {
      const int row = i >> 4, c4 = (i & 15) * 4;
      const v4f f = *(const v4f*)&x[(mBase + row) * D_MODEL + kBase + c4];
      v4bf h;
#pragma unroll
      for (int j = 0; j < 4; ++j) h[j] = (__bf16)f[j];
      *(v4bf*)&xs[buf][row * 64 + c4] = h;
    }
    // W slab transposed: wt[n][k] = W[k][n].
    for (int i = tid; i < 64 * 32; i += 256) {
      const int kl = i >> 5, n0 = (i & 31) * 4;
      const v4f f = *(const v4f*)&W[(kBase + kl) * D_HEAD + n0];
#pragma unroll
      for (int j = 0; j < 4; ++j) wt[buf][(n0 + j) * 64 + kl] = (__bf16)f[j];
    }
  };

  v8f acc[8];
#pragma unroll
  for (int t = 0; t < 8; ++t) acc[t] = v8f_zero();

  stage(0, 0);
  __syncthreads();

  for (int kb = 0; kb < D_MODEL / 64; ++kb) {
    const int cur = kb & 1;
    if (kb + 1 < D_MODEL / 64) stage((kb + 1) * 64, (kb + 1) & 1);

#pragma unroll
    for (int ks = 0; ks < 2; ++ks) {
      const v16bf a = load_frag16(xs[cur], wave * 16, ks * 32, 64);
      v16bf b[8];
#pragma unroll
      for (int t = 0; t < 8; ++t) b[t] = load_frag16(wt[cur], t * 16, ks * 32, 64);
#pragma unroll
      for (int t = 0; t < 8; ++t) acc[t] = wmma_bf16(a, b[t], acc[t]);
      SCHED_DS_WMMA(18, 8);   // 18 ds_load_b128, then 8 WMMAs, per k-step
    }
    __syncthreads();
  }

  // C/D layout: lane holds col n = lane&15; VGPR r holds row 8*(lane>>4)+r.
#pragma unroll
  for (int t = 0; t < 8; ++t)
#pragma unroll
    for (int r = 0; r < 8; ++r) {
      const int m = mBase + wave * 16 + hlf * 8 + r;
      const int n = t * 16 + rr;
      dst[m * D_HEAD + n] = (__bf16)(acc[t][r] * scale);
    }
}

// ---------------------------------------------------------------------------
// Kernel 2: causal flash attention.
// grid = 128; block = 128 (4 waves). Block owns 64 query rows; wave owns 16.
// K tile (64x128 bf16, contiguous 16 KB) staged by the Tensor Data Mover;
// V tile staged transposed by VALU. Online softmax in the C layout with
// half-group shuffles; P re-fragmented through LDS.
// ---------------------------------------------------------------------------
__global__ __launch_bounds__(128, 1) void attn_kernel(
    const __bf16* __restrict__ Qb, const __bf16* __restrict__ Kb,
    const __bf16* __restrict__ Vb, __bf16* __restrict__ Ob)
{
  __shared__ alignas(16) __bf16 Ksh[64 * 128];      // [key][d]   16 KB
  __shared__ alignas(16) __bf16 Vt [128 * 64];      // [d][key]   16 KB
  __shared__ alignas(16) __bf16 Ps [4 * 16 * 64];   // per-wave P  8 KB

  const int tid  = threadIdx.x;
  const int wave = tid >> 5;
  const int lane = tid & 31;
  const int hlf  = lane >> 4;
  const int rr   = lane & 15;
  const int qBase = blockIdx.x * 64;
  const int qRow0 = qBase + wave * 16;

  v16bf qf[4];
#pragma unroll
  for (int ks = 0; ks < 4; ++ks)
    qf[ks] = load_frag16(Qb + qRow0 * D_HEAD, 0, ks * 32, D_HEAD);

  v8f oacc[8];
#pragma unroll
  for (int t = 0; t < 8; ++t) oacc[t] = v8f_zero();
  float mrun[8], lrun[8];
#pragma unroll
  for (int r = 0; r < 8; ++r) { mrun[r] = -1.0e30f; lrun[r] = 0.0f; }

  __bf16* __restrict__ Pw = Ps + wave * (16 * 64);

#if USE_TDM
  const unsigned lds_k = (unsigned)(size_t)(&Ksh[0]); // low 32b = LDS offset
#endif

  for (int kv = 0; kv < qBase + 64; kv += 64) {
#if USE_TDM
    // --- K tile via Tensor Data Mover (D# per ISA 8.3/8.4) ---
    if (wave == 0) {
      const unsigned long long ga = (unsigned long long)(size_t)(Kb + (size_t)kv * D_HEAD);
      u32x4 g0;
      g0[0] = 1u;                                   // count=1, user mode
      g0[1] = lds_k;                                // lds_addr
      g0[2] = (unsigned)(ga & 0xffffffffu);         // global_addr[31:0]
      g0[3] = (unsigned)((ga >> 32) & 0x1ffffffu)   // global_addr[56:32]
              | (2u << 30);                         // type=2 (image)
      i32x8 g1;
      g1[0] = (int)(1u << 16);                      // data_size=2B, mask=0
      g1[1] = (int)((unsigned)D_HEAD << 16);        // tensor_dim0[15:0]=128
      g1[2] = (int)((unsigned)N_CTX << 16);         // dim0[31:16]=0 | dim1 lo=8192
      g1[3] = (int)((unsigned)D_HEAD << 16);        // dim1 hi=0 | tile_dim0=128
      g1[4] = 64;                                   // tile_dim1=64, tile_dim2=0
      g1[5] = D_HEAD;                               // tensor_dim0_stride=128
      g1[6] = 0;
      g1[7] = 0;
      i32x4 gz4;
      gz4[0] = 0; gz4[1] = 0; gz4[2] = 0; gz4[3] = 0;
      i32x8 gz8;
#pragma unroll
      for (int j = 0; j < 8; ++j) gz8[j] = 0;
      __builtin_amdgcn_tensor_load_to_lds(g0, g1, gz4, gz4, gz8, 0);
    }
#else
    // Fallback: vectorized copy of the K tile.
    for (int i = tid; i < 64 * 16; i += 128) {
      const int key = i >> 4, c = (i & 15) * 8;
      *(v8bf*)&Ksh[key * D_HEAD + c] =
          *(const v8bf*)&Kb[(size_t)(kv + key) * D_HEAD + c];
    }
#endif
    // V tile transposed: 16B reads, scattered bf16 LDS writes.
    for (int i = tid; i < 64 * 16; i += 128) {
      const int key = i >> 4, d0 = (i & 15) * 8;
      const v8bf v = *(const v8bf*)&Vb[(size_t)(kv + key) * D_HEAD + d0];
#pragma unroll
      for (int j = 0; j < 8; ++j) Vt[(d0 + j) * 64 + key] = v[j];
    }
#if USE_TDM
    if (wave == 0) __builtin_amdgcn_s_wait_tensorcnt(0);
#endif
    __syncthreads();

    // S = Q @ K^T : preload 4 B fragments per k-step, then WMMA burst.
    v8f sacc[4];
#pragma unroll
    for (int t = 0; t < 4; ++t) sacc[t] = v8f_zero();
#pragma unroll
    for (int ks = 0; ks < 4; ++ks) {
      v16bf b[4];
#pragma unroll
      for (int t = 0; t < 4; ++t) b[t] = load_frag16(Ksh, t * 16, ks * 32, D_HEAD);
#pragma unroll
      for (int t = 0; t < 4; ++t) sacc[t] = wmma_bf16(qf[ks], b[t], sacc[t]);
      SCHED_DS_WMMA(8, 4);
    }

    // Causal mask + per-row tile max.
    float tmax[8];
#pragma unroll
    for (int r = 0; r < 8; ++r) tmax[r] = -1.0e30f;
#pragma unroll
    for (int t = 0; t < 4; ++t)
#pragma unroll
      for (int r = 0; r < 8; ++r) {
        const int qr = qRow0 + hlf * 8 + r;
        const int kc = kv + t * 16 + rr;
        float s = (kc <= qr) ? sacc[t][r] : -1.0e30f;
        sacc[t][r] = s;
        tmax[r] = fmaxf(tmax[r], s);
      }
#pragma unroll
    for (int r = 0; r < 8; ++r)
#pragma unroll
      for (int m = 1; m < 16; m <<= 1)
        tmax[r] = fmaxf(tmax[r], __shfl_xor(tmax[r], m, 32));

    float corr[8], lsum[8];
#pragma unroll
    for (int r = 0; r < 8; ++r) {
      const float mn = fmaxf(mrun[r], tmax[r]);
      corr[r] = __expf(mrun[r] - mn);
      mrun[r] = mn;
      lsum[r] = 0.0f;
    }

    // P = exp(S - m); row sums; spill P to LDS for A-operand re-fragmenting.
#pragma unroll
    for (int t = 0; t < 4; ++t)
#pragma unroll
      for (int r = 0; r < 8; ++r) {
        const float p = __expf(sacc[t][r] - mrun[r]); // masked -> exp(-huge)=0
        lsum[r] += p;
        Pw[(hlf * 8 + r) * 64 + t * 16 + rr] = (__bf16)p;
      }
#pragma unroll
    for (int r = 0; r < 8; ++r) {
#pragma unroll
      for (int m = 1; m < 16; m <<= 1)
        lsum[r] += __shfl_xor(lsum[r], m, 32);
      lrun[r] = lrun[r] * corr[r] + lsum[r];
    }

    // Rescale running O, then O += P @ V (preload 8 B frags, WMMA burst).
#pragma unroll
    for (int t = 0; t < 8; ++t)
#pragma unroll
      for (int r = 0; r < 8; ++r) oacc[t][r] *= corr[r];
#pragma unroll
    for (int ks = 0; ks < 2; ++ks) {
      const v16bf a = load_frag16(Pw, 0, ks * 32, 64);
      v16bf b[8];
#pragma unroll
      for (int t = 0; t < 8; ++t) b[t] = load_frag16(Vt, t * 16, ks * 32, 64);
#pragma unroll
      for (int t = 0; t < 8; ++t) oacc[t] = wmma_bf16(a, b[t], oacc[t]);
      SCHED_DS_WMMA(18, 8);
    }
    __syncthreads();  // before next iteration overwrites Ksh/Vt
  }

  // Normalize and store bf16 attention output.
#pragma unroll
  for (int t = 0; t < 8; ++t)
#pragma unroll
    for (int r = 0; r < 8; ++r) {
      const int m = qRow0 + hlf * 8 + r;
      const int n = t * 16 + rr;
      Ob[m * D_HEAD + n] = (__bf16)(oacc[t][r] / lrun[r]);
    }
}

// ---------------------------------------------------------------------------
// Kernel 3: output projection.  Ob[8192,128] @ W_o[128,1024] -> fp32 out.
// grid = (64, 8); block = 256 (8 waves).
// ---------------------------------------------------------------------------
__global__ __launch_bounds__(256, 1) void out_proj_kernel(
    const __bf16* __restrict__ Ob, const float* __restrict__ Wo,
    float* __restrict__ out)
{
  __shared__ alignas(16) __bf16 wot[128 * 128];   // [n][k]  32 KB

  const int tid  = threadIdx.x;
  const int wave = tid >> 5;
  const int lane = tid & 31;
  const int hlf  = lane >> 4;
  const int rr   = lane & 15;
  const int mBase = blockIdx.x * 128;
  const int nBase = blockIdx.y * 128;

  for (int i = tid; i < 128 * 32; i += 256) {
    const int k = i >> 5, n0 = (i & 31) * 4;
    const v4f f = *(const v4f*)&Wo[k * D_MODEL + nBase + n0];
#pragma unroll
    for (int j = 0; j < 4; ++j) wot[(n0 + j) * 128 + k] = (__bf16)f[j];
  }
  __syncthreads();

  const int m0 = mBase + wave * 16;
  v8f acc[8];
#pragma unroll
  for (int t = 0; t < 8; ++t) acc[t] = v8f_zero();

#pragma unroll
  for (int ks = 0; ks < 4; ++ks) {
    const v16bf a = load_frag16(Ob + m0 * D_HEAD, 0, ks * 32, D_HEAD);
    v16bf b[8];
#pragma unroll
    for (int t = 0; t < 8; ++t) b[t] = load_frag16(wot, t * 16, ks * 32, 128);
#pragma unroll
    for (int t = 0; t < 8; ++t) acc[t] = wmma_bf16(a, b[t], acc[t]);
    SCHED_DS_WMMA(16, 8);
  }

#pragma unroll
  for (int t = 0; t < 8; ++t)
#pragma unroll
    for (int r = 0; r < 8; ++r)
      out[(m0 + hlf * 8 + r) * D_MODEL + nBase + t * 16 + rr] = acc[t][r];
}

// ---------------------------------------------------------------------------
extern "C" void kernel_launch(void* const* d_in, const int* in_sizes, int n_in,
                              void* d_out, int out_size, void* d_ws, size_t ws_size,
                              hipStream_t stream) {
  const float* x  = (const float*)d_in[0];
  const float* Wq = (const float*)d_in[1];
  const float* Wk = (const float*)d_in[2];
  const float* Wv = (const float*)d_in[3];
  const float* Wo = (const float*)d_in[4];
  float* out = (float*)d_out;

  __bf16* Qb = (__bf16*)d_ws;                 // [8192,128] bf16, Q * 1/sqrt(dh)
  __bf16* Kb = Qb + (size_t)N_CTX * D_HEAD;
  __bf16* Vb = Kb + (size_t)N_CTX * D_HEAD;
  __bf16* Ob = Vb + (size_t)N_CTX * D_HEAD;

  qkv_proj_kernel<<<dim3(N_CTX / 128, 3), 256, 0, stream>>>(x, Wq, Wk, Wv,
                                                            Qb, Kb, Vb);
  attn_kernel<<<dim3(N_CTX / 64), 128, 0, stream>>>(Qb, Kb, Vb, Ob);
  out_proj_kernel<<<dim3(N_CTX / 128, D_MODEL / 128), 256, 0, stream>>>(Ob, Wo,
                                                                        out);
}
